// ArrowLoraLinearLayer_58488864637465
// MI455X (gfx1250) — matrix operand order repaired and checked by reading, not compiled, not run on hardware
//
#include <hip/hip_runtime.h>
#include <hip/hip_bf16.h>

// Problem constants (match reference)
#define TOK    8192       // B*S tokens
#define FD     2048       // feature dim
#define EXP    16         // experts
#define RNK    16         // LoRA rank
#define ERD    256        // EXP*RNK
#define OD     2048       // output dim
#define WCROWS 272        // 16 prototype rows + 256 A rows (valid)
#define WCPAD  320        // padded to 20 tiles of 16 so 4 waves * 5 tiles, zero-filled

typedef __attribute__((ext_vector_type(16))) __bf16 v16bf;
typedef __attribute__((ext_vector_type(8)))  float  v8f;

__device__ __forceinline__ unsigned short f32_to_bf16(float f) {
    unsigned int u = __float_as_uint(f);
    u += 0x7FFFu + ((u >> 16) & 1u);      // round-to-nearest-even
    return (unsigned short)(u >> 16);
}

__device__ __forceinline__ unsigned int pack2bf(float a, float b) {
    return (unsigned int)f32_to_bf16(a) | ((unsigned int)f32_to_bf16(b) << 16);
}

// 16 bf16 fragment from two 16B chunks
__device__ __forceinline__ v16bf ldfrag(const unsigned short* p0, const unsigned short* p1) {
    v16bf r;
    ((uint4*)&r)[0] = *(const uint4*)p0;
    ((uint4*)&r)[1] = *(const uint4*)p1;
    return r;
}

// ---------------------------------------------------------------------------
// Kernel 0: build bf16 combined down-proj weight Wc[320][2048] =
//   [protos(16); A_all(256); zeros(48)]  and transposed up-proj
//   BmatT[2048][256], BmatT[o][e*16+r] = B_stack[e][o][r]
// ---------------------------------------------------------------------------
__global__ __launch_bounds__(256)
void k_prep(const float* __restrict__ proto,      // 16 x 2048
            const float* __restrict__ A,          // 256 x 2048 (E,R,F row-major)
            const float* __restrict__ Bst,        // 16 x 2048 x 16 (E,O,R)
            unsigned short* __restrict__ Wc,      // 320 x 2048 bf16
            unsigned short* __restrict__ BT) {    // 2048 x 256 bf16
    const int WCN = WCPAD * FD;
    const int BTN = OD * ERD;
    int idx = blockIdx.x * 256 + threadIdx.x;
    if (idx < WCN) {
        int row = idx / FD, col = idx % FD;
        float v = 0.0f;
        if (row < 16)            v = proto[row * FD + col];
        else if (row < WCROWS)   v = A[(size_t)(row - 16) * FD + col];
        Wc[idx] = f32_to_bf16(v);
    }
    int idx2 = idx - WCN;
    if (idx2 >= 0 && idx2 < BTN) {
        int o = idx2 / ERD, c = idx2 % ERD;
        int e = c >> 4, r = c & 15;
        BT[idx2] = f32_to_bf16(Bst[((size_t)e * OD + o) * RNK + r]);
    }
}

// ---------------------------------------------------------------------------
// Kernel 1: fused routing + LoRA-down.
//   One workgroup (128 thr = 4 waves) per 16-token tile.
//   [sim | z | pad] (16 x 320) = x_tile (16 x 2048, bf16) @ Wc^T via WMMA.
//   Every wave owns exactly 5 n-tiles (uniform, no exec divergence):
//   wave w -> tiles {w, w+4, w+8, w+12, w+16}.  Tile 0 = sim (wave 0).
//   Then per-token top-k + softmax -> coeff, w = coeff*z -> bf16 W[8192][256].
// ---------------------------------------------------------------------------
__global__ __launch_bounds__(128)
void k_route_z(const float* __restrict__ x,
               const unsigned short* __restrict__ Wc,
               const int* __restrict__ topk_p,
               unsigned short* __restrict__ W) {
    __shared__ unsigned short xs[16 * 40];    // 16 rows x 32 bf16, stride 40 (80B, bank-skewed)
    __shared__ float simbuf[16 * 16];
    __shared__ float coeff[16 * 16];

    const int tid  = threadIdx.x;
    const int lane = tid & 31;
    const int wave = tid >> 5;
    const int tokbase = blockIdx.x * 16;

    v8f acc[5];
#pragma unroll
    for (int j = 0; j < 5; ++j)
#pragma unroll
        for (int v = 0; v < 8; ++v) acc[j][v] = 0.0f;

    // A-fragment (16x32 bf16) lane layout: lanes 0-15 K{0..15}, 16-31 K{16..31}
    const int arow  = lane & 15;
    const int ahalf = lane >> 4;
    const unsigned short* a0 = &xs[arow * 40 + ahalf * 16];
    const unsigned short* a1 = &xs[arow * 40 + ahalf * 16 + 8];

    // B-fragment (32x16 bf16): column = lane%16, K half per lane/16
    const int bcol  = lane & 15;
    const int bkoff = (lane >> 4) * 16;
    const unsigned short* bptr[5];
#pragma unroll
    for (int j = 0; j < 5; ++j) {
        const int ntile = wave + 4 * j;          // 0..19, uniform per wave
        bptr[j] = &Wc[(size_t)(ntile * 16 + bcol) * FD + bkoff];
    }

    // x staging: each thread loads 4 f32 (row tid/8, cols (tid%8)*4..)
    const int sr = tid >> 3;
    const int sc = (tid & 7) * 4;

    for (int k0 = 0; k0 < FD; k0 += 32) {
        float4 xv = *(const float4*)&x[(size_t)(tokbase + sr) * FD + k0 + sc];
        uint2 p;
        p.x = pack2bf(xv.x, xv.y);
        p.y = pack2bf(xv.z, xv.w);
        *(uint2*)&xs[sr * 40 + sc] = p;
        __syncthreads();

        v16bf afrag = ldfrag(a0, a1);
        v16bf bfrag[5];
#pragma unroll
        for (int j = 0; j < 5; ++j)
            bfrag[j] = ldfrag(bptr[j] + k0, bptr[j] + k0 + 8);
#pragma unroll
        for (int j = 0; j < 5; ++j)
            acc[j] = __builtin_amdgcn_wmma_f32_16x16x32_bf16(
                false, afrag, false, bfrag[j], (short)0, acc[j], false, false);
        __syncthreads();
    }

    // C/D layout: N = lane%16, M = vgpr + 8*(lane/16)
    const int crow0 = (lane >> 4) * 8;
    const int ccol  = lane & 15;

    if (wave == 0) {                       // wave 0 owns n-tile 0 == sim
#pragma unroll
        for (int v = 0; v < 8; ++v)
            simbuf[(crow0 + v) * 16 + ccol] = fabsf(acc[0][v]);
    }
    __syncthreads();

    if (tid < 16) {                        // one thread per token: top-k + softmax
        int kk = topk_p[0];
        if (kk < 1) kk = 1; if (kk > 16) kk = 16;
        float vals[16];
#pragma unroll
        for (int e = 0; e < 16; ++e) vals[e] = simbuf[tid * 16 + e];
        int   chosen[16];
        float chval[16];
        for (int j = 0; j < kk; ++j) {
            int bi = 0; float bv = vals[0];
#pragma unroll
            for (int e = 1; e < 16; ++e)
                if (vals[e] > bv) { bv = vals[e]; bi = e; }
            chosen[j] = bi; chval[j] = bv; vals[bi] = -INFINITY;
        }
        float mx = chval[0];               // first pick is the max
        float s = 0.0f, ex[16];
        for (int j = 0; j < kk; ++j) { ex[j] = __expf(chval[j] - mx); s += ex[j]; }
        float inv = 1.0f / s;
#pragma unroll
        for (int e = 0; e < 16; ++e) coeff[tid * 16 + e] = 0.0f;
        for (int j = 0; j < kk; ++j) coeff[tid * 16 + chosen[j]] = ex[j] * inv;
    }
    __syncthreads();

    // w = coeff * z  -> bf16 W[token][256]; skip sim tile (0) and pad tiles (>16)
#pragma unroll
    for (int j = 0; j < 5; ++j) {
        const int ntile = wave + 4 * j;
        if (ntile >= 1 && ntile <= 16) {
            const int e   = ntile - 1;
            const int col = e * 16 + ccol;
#pragma unroll
            for (int v = 0; v < 8; ++v) {
                const int m = crow0 + v;
                float val = acc[j][v] * coeff[m * 16 + e];
                W[(size_t)(tokbase + m) * ERD + col] = f32_to_bf16(val);
            }
        }
    }
}

// ---------------------------------------------------------------------------
// Kernel 2: delta = scaling * (W[8192x256] @ BmatT^T) -> f32 out[8192x2048]
//   256 thr = 8 waves; each wave computes 4 adjacent 16x16 tiles (A-frag
//   reused 4x); K = 256 fully unrolled -> 32 WMMA per wave.
// ---------------------------------------------------------------------------
__global__ __launch_bounds__(256)
void k_up_gemm(const unsigned short* __restrict__ W,
               const unsigned short* __restrict__ BT,
               const float* __restrict__ scaling,
               float* __restrict__ out) {
    const int tid  = threadIdx.x;
    const int lane = tid & 31;
    const int wave = tid >> 5;
    const int ntile0 = (blockIdx.x * 8 + wave) * 4;  // 0..124
    const int mtile  = blockIdx.y;                   // 0..511

    const int arow  = lane & 15;
    const int ahalf = lane >> 4;
    const size_t abase = (size_t)(mtile * 16 + arow) * ERD + ahalf * 16;

    const unsigned short* bptr[4];
#pragma unroll
    for (int j = 0; j < 4; ++j)
        bptr[j] = &BT[(size_t)((ntile0 + j) * 16 + (lane & 15)) * ERD + (lane >> 4) * 16];

    v8f acc[4];
#pragma unroll
    for (int j = 0; j < 4; ++j)
#pragma unroll
        for (int v = 0; v < 8; ++v) acc[j][v] = 0.0f;

#pragma unroll
    for (int k0 = 0; k0 < ERD; k0 += 32) {
        v16bf a = ldfrag(&W[abase + k0], &W[abase + k0 + 8]);
        v16bf b[4];
#pragma unroll
        for (int j = 0; j < 4; ++j)
            b[j] = ldfrag(bptr[j] + k0, bptr[j] + k0 + 8);
#pragma unroll
        for (int j = 0; j < 4; ++j)
            acc[j] = __builtin_amdgcn_wmma_f32_16x16x32_bf16(
                false, a, false, b[j], (short)0, acc[j], false, false);
    }

    const float s = scaling[0];
    const int crow0 = (lane >> 4) * 8;
    const int ccol  = lane & 15;
#pragma unroll
    for (int j = 0; j < 4; ++j)
#pragma unroll
        for (int v = 0; v < 8; ++v)
            out[(size_t)(mtile * 16 + crow0 + v) * OD + (ntile0 + j) * 16 + ccol]
                = acc[j][v] * s;
}

// ---------------------------------------------------------------------------
extern "C" void kernel_launch(void* const* d_in, const int* in_sizes, int n_in,
                              void* d_out, int out_size, void* d_ws, size_t ws_size,
                              hipStream_t stream) {
    const float* x     = (const float*)d_in[0];   // 8192 x 2048
    const float* proto = (const float*)d_in[1];   // 16 x 2048
    const float* Ast   = (const float*)d_in[2];   // 16 x 16 x 2048
    const float* Bst   = (const float*)d_in[3];   // 16 x 2048 x 16
    const float* scal  = (const float*)d_in[4];   // scalar
    const int*   topk  = (const int*)d_in[5];     // scalar
    float* out = (float*)d_out;

    // Workspace layout (bf16), all 16B-aligned:
    unsigned short* Wc = (unsigned short*)d_ws;           // 320*2048
    unsigned short* BT = Wc + (size_t)WCPAD * FD;         // 2048*256
    unsigned short* W  = BT + (size_t)OD * ERD;           // 8192*256
    // total ~6.6 MB

    {   // prep: (320*2048 + 2048*256) elements
        int total = WCPAD * FD + OD * ERD;
        int blocks = (total + 255) / 256;
        k_prep<<<blocks, 256, 0, stream>>>(proto, Ast, Bst, Wc, BT);
    }
    {   // fused routing + down-proj: one WG per 16-token tile
        k_route_z<<<TOK / 16, 128, 0, stream>>>(x, Wc, topk, W);
    }
    {   // up-proj GEMM: 4 n-tiles per wave, 8 waves -> 512 cols per block
        dim3 grid(OD / (16 * 8 * 4), TOK / 16);
        k_up_gemm<<<grid, 256, 0, stream>>>(W, BT, scal, out);
    }
}
